// tcNet_51316269253249
// MI455X (gfx1250) — compile-verified, matrix operand order checked by reading, not run on hardware
//
#include <hip/hip_runtime.h>
#include <hip/hip_bf16.h>
#include <math.h>

// ---------------------------------------------------------------------------
// Problem constants (from reference setup_inputs)
// ---------------------------------------------------------------------------
#define B_   16
#define T_   128
#define DIN_ 1024
#define H_   1024
#define G_   2048   // 2*H (gate width)
#define KG_  2048   // DIN + H (GRU concat K)
#define D_   1024
#define E_   25
#define C_   4716
#define CT_  295    // ceil(C/16)

typedef __attribute__((ext_vector_type(16))) __bf16 v16bf;
typedef __attribute__((ext_vector_type(8)))  float  v8f;

__device__ __forceinline__ v8f zero8() {
  v8f v;
#pragma unroll
  for (int i = 0; i < 8; ++i) v[i] = 0.0f;
  return v;
}

__device__ __forceinline__ v8f wmma_bf16(v16bf a, v16bf b, v8f c) {
  return __builtin_amdgcn_wmma_f32_16x16x32_bf16(
      false, a, false, b, (short)0, c, false, false);
}

__device__ __forceinline__ const v16bf* fragp(const __bf16* p) {
  return (const v16bf*)p;
}

// Fragment layout (ISA 7.12.2, 16-bit A 16x32 / B 32x16):
//   lane = hi*16 + (row|col),  hi selects K offset +8
//   elems 0..7  <-> K = ktile*32 + hi*8 + 0..7
//   elems 8..15 <-> K = ktile*32 + hi*8 + 16..23
// Inverse map: (m, k) -> flat half index inside an A-frag buffer.
__device__ __forceinline__ int afrag_idx(int m, int k) {
  const int kt = k >> 5, koff = k & 31;
  const int hi   = (koff >> 3) & 1;  // [8,16) or [24,32)
  const int grp  = koff >> 4;        // 0: k<16, 1: k>=16
  const int elem = grp * 8 + (koff & 7);
  const int lane = hi * 16 + m;
  return (kt * 32 + lane) * 16 + elem;
}

// ---------------------------------------------------------------------------
// Swizzle kernels (one-time / cheap data-layout passes)
// ---------------------------------------------------------------------------
// B-matrix (K x N, fp32 row-major) -> bf16 fragments [(nt*KT+kt)*32+lane]
__global__ void k_swz_b(const float* __restrict__ src, __bf16* __restrict__ dst,
                        int K, int N) {
  const int i = blockIdx.x * blockDim.x + threadIdx.x;
  const int KT = K >> 5;
  if (i >= KT * (N >> 4) * 32) return;
  const int lane = i & 31;
  const int kt = (i >> 5) % KT;
  const int nt = (i >> 5) / KT;
  const int n = nt * 16 + (lane & 15);
  const int base = (lane >> 4) * 8;
  const int k0 = kt * 32 + base;
  v16bf v;
#pragma unroll
  for (int q = 0; q < 8; ++q) {
    v[q]     = (__bf16)src[(size_t)(k0 + q) * N + n];
    v[8 + q] = (__bf16)src[(size_t)(k0 + 16 + q) * N + n];
  }
  ((v16bf*)dst)[i] = v;
}

// GRU x-operand: frame_features -> per-(dir,t) A-fragments (bw reversal folded in)
__global__ void k_swz_x(const float* __restrict__ x, const int* __restrict__ lens,
                        __bf16* __restrict__ dfw, __bf16* __restrict__ dbw) {
  const int i = blockIdx.x * blockDim.x + threadIdx.x;
  const int per = T_ * 32 * 32;  // 32 ktiles * 32 lanes per step
  if (i >= 2 * per) return;
  const int dir = i / per;
  const int r = i % per;
  const int lane = r & 31;
  const int kt = (r >> 5) & 31;
  const int t = r >> 10;
  const int m = lane & 15;
  const int base = (lane >> 4) * 8;
  const int L = lens[m];
  const int tt = dir ? ((t < L) ? (L - 1 - t) : t) : t;
  const float* row = x + ((size_t)m * T_ + tt) * DIN_ + kt * 32 + base;
  v16bf v;
#pragma unroll
  for (int q = 0; q < 8; ++q) { v[q] = (__bf16)row[q]; v[8 + q] = (__bf16)row[16 + q]; }
  ((v16bf*)(dir ? dbw : dfw))[r] = v;
}

// 16 x K fp32 row-major -> A fragments (cnn_out, cls_feat)
__global__ void k_swz_a16(const float* __restrict__ src, __bf16* __restrict__ dst, int K) {
  const int i = blockIdx.x * blockDim.x + threadIdx.x;
  if (i >= (K >> 5) * 32) return;
  const int lane = i & 31, kt = i >> 5;
  const int m = lane & 15, base = (lane >> 4) * 8;
  const float* row = src + (size_t)m * K + kt * 32 + base;
  v16bf v;
#pragma unroll
  for (int q = 0; q < 8; ++q) { v[q] = (__bf16)row[q]; v[8 + q] = (__bf16)row[16 + q]; }
  ((v16bf*)dst)[i] = v;
}

// im2col A fragments for width-3 SAME conv: rows = (b, t), K = 3*1024
__global__ void k_swz_im2col(const float* __restrict__ in, __bf16* __restrict__ dst,
                             int Tlen, int total) {
  const int i = blockIdx.x * blockDim.x + threadIdx.x;
  if (i >= total) return;
  const int lane = i & 31;
  const int kt = (i >> 5) % 96;
  const int mt = (i >> 5) / 96;
  const int m = lane & 15, base = (lane >> 4) * 8;
  const int row = mt * 16 + m;
  const int b = row / Tlen, tp = row % Tlen;
  const int k0 = kt * 32 + base;      // both 8-chunks stay in one dt slice
  const int dt = k0 >> 10, ci = k0 & 1023;
  const int ti = tp + dt - 1;
  v16bf v;
  if (ti >= 0 && ti < Tlen) {
    const float* p = in + ((size_t)b * Tlen + ti) * 1024 + ci;
#pragma unroll
    for (int q = 0; q < 8; ++q) { v[q] = (__bf16)p[q]; v[8 + q] = (__bf16)p[16 + q]; }
  } else {
#pragma unroll
    for (int q = 0; q < 16; ++q) v[q] = (__bf16)0.0f;
  }
  ((v16bf*)dst)[i] = v;
}

// ---------------------------------------------------------------------------
// Utility kernels
// ---------------------------------------------------------------------------
__global__ void k_zero(float* __restrict__ p, int n) {
  int i = blockIdx.x * blockDim.x + threadIdx.x;
  if (i < n) p[i] = 0.0f;
}

__global__ void k_add(const float* __restrict__ a, const float* __restrict__ b,
                      float* __restrict__ o, int n) {
  int i = blockIdx.x * blockDim.x + threadIdx.x;
  if (i < n) o[i] = a[i] + b[i];
}

// ---------------------------------------------------------------------------
// GRU step: gates  g = sigmoid([x_t, h] @ Wg + bg)
// grid = (G_/16, 2 dirs), block = 32 (one wave = one 16x16 tile)
// Epilogue also emits the r*h A-fragment (bf16) for the candidate GEMM.
// ---------------------------------------------------------------------------
__global__ void k_gru_gates(const __bf16* __restrict__ xfrag_fw, const __bf16* __restrict__ xfrag_bw,
                            const __bf16* __restrict__ hfrag_fw, const __bf16* __restrict__ hfrag_bw,
                            const __bf16* __restrict__ wg_fw, const __bf16* __restrict__ wg_bw,
                            const float* __restrict__ bg_fw, const float* __restrict__ bg_bw,
                            const float* __restrict__ hin_fw, const float* __restrict__ hin_bw,
                            float* __restrict__ gates_fw, float* __restrict__ gates_bw,
                            __bf16* __restrict__ rhfrag_fw, __bf16* __restrict__ rhfrag_bw,
                            int t) {
  const int dir = blockIdx.y, lane = threadIdx.x, nt = blockIdx.x;
  const v16bf* xa = fragp(dir ? xfrag_bw : xfrag_fw) + (size_t)t * (32 * 32);
  const v16bf* ha = fragp(dir ? hfrag_bw : hfrag_fw);
  const v16bf* wb = fragp(dir ? wg_bw : wg_fw) + (size_t)nt * (64 * 32);

  v8f acc = zero8();
#pragma unroll 4
  for (int kt = 0; kt < 32; ++kt)
    acc = wmma_bf16(xa[kt * 32 + lane], wb[kt * 32 + lane], acc);
#pragma unroll 4
  for (int kt = 0; kt < 32; ++kt)
    acc = wmma_bf16(ha[kt * 32 + lane], wb[(32 + kt) * 32 + lane], acc);

  const float* bg  = dir ? bg_bw : bg_fw;
  const float* hin = dir ? hin_bw : hin_fw;
  float* gates     = dir ? gates_bw : gates_fw;
  __bf16* rhf      = dir ? rhfrag_bw : rhfrag_fw;

  const int m = lane & 15, ncol = nt * 16 + m, mo = (lane >> 4) * 8;
  const float bias = bg[ncol];
#pragma unroll
  for (int r = 0; r < 8; ++r) {
    const int mr = r + mo;
    const float g = 1.0f / (1.0f + expf(-(acc[r] + bias)));
    gates[(size_t)mr * G_ + ncol] = g;
    if (ncol < H_) {  // r-gate columns: emit r*h fragment for candidate GEMM
      const float rh = g * hin[(size_t)mr * H_ + ncol];
      rhf[afrag_idx(mr, ncol)] = (__bf16)rh;
    }
  }
}

// ---------------------------------------------------------------------------
// GRU step: candidate + state update
//   c = tanh([x_t, r*h] @ Wc + bc);  hn = z*h + (1-z)*c
// grid = (H_/16, 2 dirs); h double-buffered; also writes next h A-fragment.
// ---------------------------------------------------------------------------
__global__ void k_gru_update(const __bf16* __restrict__ xfrag_fw, const __bf16* __restrict__ xfrag_bw,
                             const __bf16* __restrict__ rhfrag_fw, const __bf16* __restrict__ rhfrag_bw,
                             const __bf16* __restrict__ wc_fw, const __bf16* __restrict__ wc_bw,
                             const float* __restrict__ bc_fw, const float* __restrict__ bc_bw,
                             const float* __restrict__ hin_fw, const float* __restrict__ hin_bw,
                             float* __restrict__ hout_fw, float* __restrict__ hout_bw,
                             __bf16* __restrict__ hfo_fw, __bf16* __restrict__ hfo_bw,
                             const float* __restrict__ gates_fw, const float* __restrict__ gates_bw,
                             float* __restrict__ ys_fw, float* __restrict__ ys_bw,
                             const int* __restrict__ lens, int t) {
  const int dir = blockIdx.y, lane = threadIdx.x, nt = blockIdx.x;
  const v16bf* xa = fragp(dir ? xfrag_bw : xfrag_fw) + (size_t)t * (32 * 32);
  const v16bf* ra = fragp(dir ? rhfrag_bw : rhfrag_fw);
  const v16bf* wb = fragp(dir ? wc_bw : wc_fw) + (size_t)nt * (64 * 32);

  v8f acc = zero8();
#pragma unroll 4
  for (int kt = 0; kt < 32; ++kt)
    acc = wmma_bf16(xa[kt * 32 + lane], wb[kt * 32 + lane], acc);
#pragma unroll 4
  for (int kt = 0; kt < 32; ++kt)
    acc = wmma_bf16(ra[kt * 32 + lane], wb[(32 + kt) * 32 + lane], acc);

  const float* bc  = dir ? bc_bw : bc_fw;
  const float* hin = dir ? hin_bw : hin_fw;
  float* hout      = dir ? hout_bw : hout_fw;
  __bf16* hfo      = dir ? hfo_bw : hfo_fw;
  const float* gts = dir ? gates_bw : gates_fw;
  float* ys        = dir ? ys_bw : ys_fw;

  const int m = lane & 15, ncol = nt * 16 + m, mo = (lane >> 4) * 8;
  const float bias = bc[ncol];
#pragma unroll
  for (int r = 0; r < 8; ++r) {
    const int mr = r + mo;
    const float c  = tanhf(acc[r] + bias);
    const float z  = gts[(size_t)mr * G_ + H_ + ncol];
    const float h0 = hin[(size_t)mr * H_ + ncol];
    const float hn = z * h0 + (1.0f - z) * c;
    const int Lr = lens[mr];
    const bool mk = (t < Lr);
    const float hs = mk ? hn : h0;
    hout[(size_t)mr * H_ + ncol] = hs;
    hfo[afrag_idx(mr, ncol)] = (__bf16)hs;
    const int ot = dir ? ((t < Lr) ? (Lr - 1 - t) : t) : t;
    ys[((size_t)mr * T_ + ot) * H_ + ncol] = mk ? hn : 0.0f;
  }
}

// ---------------------------------------------------------------------------
// width-3 SAME conv as GEMM on pre-swizzled im2col fragments, fused ReLU
// grid = (B*Tlen/16, 64), block = 32, KT = 96
// ---------------------------------------------------------------------------
__global__ void k_conv(const __bf16* __restrict__ afrag, const __bf16* __restrict__ wswz,
                       float* __restrict__ out, int Tlen) {
  const int lane = threadIdx.x, mt = blockIdx.x, nt = blockIdx.y;
  const v16bf* aa = fragp(afrag) + (size_t)mt * (96 * 32);
  const v16bf* bb = fragp(wswz) + (size_t)nt * (96 * 32);
  v8f acc = zero8();
#pragma unroll 4
  for (int kt = 0; kt < 96; ++kt)
    acc = wmma_bf16(aa[kt * 32 + lane], bb[kt * 32 + lane], acc);
  const int m = lane & 15, ncol = nt * 16 + m, mo = (lane >> 4) * 8;
#pragma unroll
  for (int r = 0; r < 8; ++r) {
    const int mr = mt * 16 + r + mo;
    const int bo = mr / Tlen, to = mr % Tlen;
    const float v = acc[r];
    out[((size_t)bo * Tlen + to) * 1024 + ncol] = v > 0.0f ? v : 0.0f;
  }
}

__global__ void k_maxpool(const float* __restrict__ in, float* __restrict__ out, int n) {
  int i = blockIdx.x * blockDim.x + threadIdx.x;
  if (i < n) {
    const int c = i & 1023, tp = (i >> 10) & 63, b = i >> 16;
    out[i] = fmaxf(in[((size_t)b * T_ + 2 * tp) * 1024 + c],
                   in[((size_t)b * T_ + 2 * tp + 1) * 1024 + c]);
  }
}

__global__ void k_maxt(const float* __restrict__ in, float* __restrict__ out, int Tlen) {
  int i = blockIdx.x * blockDim.x + threadIdx.x;
  if (i < B_ * 1024) {
    const int c = i & 1023, b = i >> 10;
    float mv = in[((size_t)b * Tlen) * 1024 + c];
    for (int t = 1; t < Tlen; ++t)
      mv = fmaxf(mv, in[((size_t)b * Tlen + t) * 1024 + c]);
    out[i] = mv;
  }
}

// cls_feat = relu(cnn_out @ fcf_W + fcf_b): grid = 64 N tiles, KT = 32
__global__ void k_fcf(const __bf16* __restrict__ afrag, const __bf16* __restrict__ wswz,
                      const float* __restrict__ bias, float* __restrict__ out) {
  const int lane = threadIdx.x, nt = blockIdx.x;
  const v16bf* aa = fragp(afrag);
  const v16bf* bb = fragp(wswz) + (size_t)nt * (32 * 32);
  v8f acc = zero8();
#pragma unroll 4
  for (int kt = 0; kt < 32; ++kt)
    acc = wmma_bf16(aa[kt * 32 + lane], bb[kt * 32 + lane], acc);
  const int m = lane & 15, ncol = nt * 16 + m, mo = (lane >> 4) * 8;
  const float bi = bias[ncol];
#pragma unroll
  for (int r = 0; r < 8; ++r) {
    const float v = acc[r] + bi;
    out[(size_t)(r + mo) * D_ + ncol] = v > 0.0f ? v : 0.0f;
  }
}

// logits1 softmax gate (tiny, VALU)
__global__ void k_cls1_gate(const float* __restrict__ feat, const float* __restrict__ w,
                            const float* __restrict__ bias, float* __restrict__ gate) {
  __shared__ float lg[B_][E_];
  const int tid = threadIdx.x;
  if (tid < B_ * E_) {
    const int b = tid / E_, e = tid % E_;
    float s = bias[e];
    for (int k = 0; k < D_; ++k) s += feat[b * D_ + k] * w[k * E_ + e];
    lg[b][e] = s;
  }
  __syncthreads();
  if (tid < B_ * E_) {
    const int b = tid / E_, e = tid % E_;
    float mx = lg[b][0];
    for (int j = 1; j < E_; ++j) mx = fmaxf(mx, lg[b][j]);
    float sum = 0.0f;
    for (int j = 0; j < E_; ++j) sum += expf(lg[b][j] - mx);
    gate[e * B_ + b] = expf(lg[b][e] - mx) / sum;
  }
}

// lvl2 einsum + gated sigmoid mixture + logit transform.
// classifiers (483 MB) read fp32 exactly once -> convert in-register (HBM floor).
__global__ void k_final(const __bf16* __restrict__ afrag, const float* __restrict__ cw,
                        const float* __restrict__ gate, float* __restrict__ out) {
  const int lane = threadIdx.x, ct = blockIdx.x;
  const int m = lane & 15;
  const int c = ct * 16 + m;
  const int base = (lane >> 4) * 8;
  const int mo = (lane >> 4) * 8;
  const v16bf* aa = fragp(afrag);

  float accum[8] = {0.f, 0.f, 0.f, 0.f, 0.f, 0.f, 0.f, 0.f};
  for (int e = 0; e < E_; ++e) {
    const float* we = cw + (size_t)e * D_ * C_;
    v8f acc = zero8();
    for (int kt = 0; kt < 32; ++kt) {
      const v16bf a = aa[kt * 32 + lane];
      v16bf b;
      const int k0 = kt * 32 + base;
#pragma unroll
      for (int q = 0; q < 8; ++q) {
        b[q]     = (__bf16)((c < C_) ? we[(size_t)(k0 + q) * C_ + c] : 0.0f);
        b[8 + q] = (__bf16)((c < C_) ? we[(size_t)(k0 + 16 + q) * C_ + c] : 0.0f);
      }
      acc = wmma_bf16(a, b, acc);
    }
#pragma unroll
    for (int r = 0; r < 8; ++r) {
      const float sg = 1.0f / (1.0f + expf(-acc[r]));
      accum[r] += gate[e * B_ + (r + mo)] * sg;
    }
  }
  if (c < C_) {
#pragma unroll
    for (int r = 0; r < 8; ++r) {
      const float p = accum[r];
      out[(size_t)(r + mo) * C_ + c] = logf(p / (1.0f - p));
    }
  }
}

// ---------------------------------------------------------------------------
// Host orchestration
// ---------------------------------------------------------------------------
extern "C" void kernel_launch(void* const* d_in, const int* in_sizes, int n_in,
                              void* d_out, int out_size, void* d_ws, size_t ws_size,
                              hipStream_t stream) {
  (void)in_sizes; (void)n_in; (void)out_size; (void)ws_size;

  char* wsb = (char*)d_ws;
  size_t off = 0;
  auto alloc = [&](size_t bytes) -> void* {
    off = (off + 255) & ~(size_t)255;
    void* p = wsb + off;
    off += bytes;
    return p;
  };

  // swizzled bf16 weight fragments
  __bf16* wg_fw = (__bf16*)alloc((size_t)KG_ * G_ * 2);
  __bf16* wg_bw = (__bf16*)alloc((size_t)KG_ * G_ * 2);
  __bf16* wc_fw = (__bf16*)alloc((size_t)KG_ * H_ * 2);
  __bf16* wc_bw = (__bf16*)alloc((size_t)KG_ * H_ * 2);
  __bf16* c0w   = (__bf16*)alloc((size_t)3 * 1024 * 1024 * 2);
  __bf16* c1w   = (__bf16*)alloc((size_t)3 * 1024 * 1024 * 2);
  __bf16* fcw   = (__bf16*)alloc((size_t)1024 * 1024 * 2);

  // pre-swizzled x operand fragments (per dir, per t)
  __bf16* xf_fw = (__bf16*)alloc((size_t)T_ * 32 * 32 * 16 * 2);
  __bf16* xf_bw = (__bf16*)alloc((size_t)T_ * 32 * 32 * 16 * 2);

  // recurrent fragment buffers (zero-init region: 4 h-frag ping-pong + 2 rh)
  const size_t hf_elems = (size_t)32 * 32 * 16;          // 16K halves = 32 KB
  __bf16* hfrags = (__bf16*)alloc(6 * hf_elems * 2);     // fw0,fw1,bw0,bw1,rh_fw,rh_bw
  __bf16* hf_fw[2] = {hfrags, hfrags + hf_elems};
  __bf16* hf_bw[2] = {hfrags + 2 * hf_elems, hfrags + 3 * hf_elems};
  __bf16* rh_fw = hfrags + 4 * hf_elems;
  __bf16* rh_bw = hfrags + 5 * hf_elems;

  // fp32 state / intermediates
  float* hbuf     = (float*)alloc((size_t)4 * B_ * H_ * 4);
  float* gates_fw = (float*)alloc((size_t)B_ * G_ * 4);
  float* gates_bw = (float*)alloc((size_t)B_ * G_ * 4);
  float* ys_fw    = (float*)alloc((size_t)B_ * T_ * H_ * 4);
  float* ys_bw    = (float*)alloc((size_t)B_ * T_ * H_ * 4);
  float* feats    = (float*)alloc((size_t)B_ * T_ * H_ * 4);
  __bf16* im0     = (__bf16*)alloc((size_t)(B_ * T_ / 16) * 96 * 32 * 16 * 2);
  __bf16* im1     = (__bf16*)alloc((size_t)(B_ * 64 / 16) * 96 * 32 * 16 * 2);
  float* c0out    = (float*)alloc((size_t)B_ * T_ * 1024 * 4);
  float* pooled   = (float*)alloc((size_t)B_ * 64 * 1024 * 4);
  float* c1out    = (float*)alloc((size_t)B_ * 64 * 1024 * 4);
  float* cnn      = (float*)alloc((size_t)B_ * 1024 * 4);
  float* clsf     = (float*)alloc((size_t)B_ * 1024 * 4);
  __bf16* af_cnn  = (__bf16*)alloc((size_t)32 * 32 * 16 * 2);
  __bf16* af_cls  = (__bf16*)alloc((size_t)32 * 32 * 16 * 2);
  float* gate     = (float*)alloc((size_t)E_ * B_ * 4);

  auto swzb = [&](const void* src, __bf16* dst, int K, int N) {
    const int tot = (K >> 5) * (N >> 4) * 32;
    k_swz_b<<<(tot + 255) / 256, 256, 0, stream>>>((const float*)src, dst, K, N);
  };
  swzb(d_in[1],  wg_fw, KG_, G_);
  swzb(d_in[5],  wg_bw, KG_, G_);
  swzb(d_in[3],  wc_fw, KG_, H_);
  swzb(d_in[7],  wc_bw, KG_, H_);
  swzb(d_in[9],  c0w, 3 * 1024, 1024);
  swzb(d_in[10], c1w, 3 * 1024, 1024);
  swzb(d_in[11], fcw, 1024, 1024);

  const float* x  = (const float*)d_in[0];
  const int* lens = (const int*)d_in[16];

  {
    const int tot = 2 * T_ * 32 * 32;
    k_swz_x<<<(tot + 255) / 256, 256, 0, stream>>>(x, lens, xf_fw, xf_bw);
  }

  // zero recurrent state (fp32 h ping-pong + bf16 h fragments)
  k_zero<<<(4 * B_ * H_ + 255) / 256, 256, 0, stream>>>(hbuf, 4 * B_ * H_);
  k_zero<<<((int)(6 * hf_elems / 2) + 255) / 256, 256, 0, stream>>>(
      (float*)hfrags, (int)(6 * hf_elems / 2));

  float* hfw[2] = {hbuf, hbuf + (size_t)B_ * H_};
  float* hbw[2] = {hbuf + (size_t)2 * B_ * H_, hbuf + (size_t)3 * B_ * H_};

  for (int t = 0; t < T_; ++t) {
    const int cur = t & 1, nxt = cur ^ 1;
    k_gru_gates<<<dim3(G_ / 16, 2), 32, 0, stream>>>(
        xf_fw, xf_bw, hf_fw[cur], hf_bw[cur], wg_fw, wg_bw,
        (const float*)d_in[2], (const float*)d_in[6],
        hfw[cur], hbw[cur], gates_fw, gates_bw, rh_fw, rh_bw, t);
    k_gru_update<<<dim3(H_ / 16, 2), 32, 0, stream>>>(
        xf_fw, xf_bw, rh_fw, rh_bw, wc_fw, wc_bw,
        (const float*)d_in[4], (const float*)d_in[8],
        hfw[cur], hbw[cur], hfw[nxt], hbw[nxt], hf_fw[nxt], hf_bw[nxt],
        gates_fw, gates_bw, ys_fw, ys_bw, lens, t);
  }

  const int nf = B_ * T_ * H_;
  k_add<<<(nf + 255) / 256, 256, 0, stream>>>(ys_fw, ys_bw, feats, nf);

  {
    const int tot = (B_ * T_ / 16) * 96 * 32;
    k_swz_im2col<<<(tot + 255) / 256, 256, 0, stream>>>(feats, im0, T_, tot);
  }
  k_conv<<<dim3(B_ * T_ / 16, 64), 32, 0, stream>>>(im0, c0w, c0out, T_);

  const int np = B_ * 64 * 1024;
  k_maxpool<<<(np + 255) / 256, 256, 0, stream>>>(c0out, pooled, np);

  {
    const int tot = (B_ * 64 / 16) * 96 * 32;
    k_swz_im2col<<<(tot + 255) / 256, 256, 0, stream>>>(pooled, im1, 64, tot);
  }
  k_conv<<<dim3(B_ * 64 / 16, 64), 32, 0, stream>>>(im1, c1w, c1out, 64);

  k_maxt<<<(B_ * 1024 + 255) / 256, 256, 0, stream>>>(c1out, cnn, 64);

  k_swz_a16<<<(1024 + 255) / 256, 256, 0, stream>>>(cnn, af_cnn, 1024);
  k_fcf<<<64, 32, 0, stream>>>(af_cnn, fcw, (const float*)d_in[12], clsf);

  k_cls1_gate<<<1, 512, 0, stream>>>(clsf, (const float*)d_in[13],
                                     (const float*)d_in[14], gate);

  k_swz_a16<<<(1024 + 255) / 256, 256, 0, stream>>>(clsf, af_cls, 1024);
  k_final<<<CT_, 32, 0, stream>>>(af_cls, (const float*)d_in[15], gate, (float*)d_out);
}
